// PubmedGAT_46076409152404
// MI455X (gfx1250) — matrix operand order repaired and checked by reading, not compiled, not run on hardware
//
#include <hip/hip_runtime.h>
#include <hip/hip_bf16.h>
#include <math.h>

// ---------------- problem constants (from reference) ----------------
#define N_NODES   20000
#define N_EDGES   320000
#define N_EDGES_SL (N_EDGES + N_NODES)   // 340000 with self loops
#define HEADS     8
#define DIM_IN    500
#define KPAD1     512                    // DIM_IN padded to multiple of 32
#define DIM_H     32
#define DIM_OUT   16
#define H1        (HEADS * DIM_H)    // 256
#define H2        (HEADS * DIM_OUT)  // 128
#define NEG_SLOPE 0.2f
#define NTG       4                  // n-tiles per wave in the WMMA GEMM

typedef __attribute__((ext_vector_type(16))) _Float16 v16h;
typedef __attribute__((ext_vector_type(8)))  _Float16 v8h;
typedef __attribute__((ext_vector_type(8)))  float    v8f;

// ---------------- helpers ----------------
__device__ inline void edge_sd(const int* __restrict__ ei, int e, int& s, int& d) {
    if (e < N_EDGES) { s = ei[e]; d = ei[N_EDGES + e]; }
    else             { s = e - N_EDGES; d = s; }
}

__device__ inline void atomicMaxF(float* addr, float val) {
    unsigned int* ua = (unsigned int*)addr;
    unsigned int old = *(volatile unsigned int*)ua;
    while (__uint_as_float(old) < val) {
        unsigned int assumed = old;
        old = atomicCAS(ua, assumed, __float_as_uint(val));
        if (old == assumed) break;
    }
}

// ---------------- f32 -> f16 convert with row padding ----------------
// dst[r*Ks + k] = (k < K) ? (f16)src[r*K + k] : 0
__global__ void convert_f16_pad(const float* __restrict__ src, _Float16* __restrict__ dst,
                                int K, int Ks, long rows)
{
    long t = (long)blockIdx.x * blockDim.x + threadIdx.x;
    if (t >= rows * Ks) return;
    long r = t / Ks;
    int  k = (int)(t - r * Ks);
    dst[t] = (k < K) ? (_Float16)src[r * K + k] : (_Float16)0.0f;
}

// ---------------- WMMA GEMM: Y[M,Nout] = Xh[M,Ks] * Wh[Nout,Ks]^T + bias ----------------
// Inputs pre-converted to f16 with zero-padded row stride Ks (multiple of 32).
// One wave32 computes a 16 x (16*NTG) strip: A fragment loaded once per k-step,
// reused across NTG B fragments -> NTG back-to-back v_wmma per iteration.
__global__ __launch_bounds__(256) void wmma_gemm_xWT(
    const _Float16* __restrict__ Xh, const _Float16* __restrict__ Wh,
    const float* __restrict__ bias, float* __restrict__ Y,
    int M, int Ks, int Nout)
{
    const int lane   = threadIdx.x & 31;
    const int wave   = threadIdx.x >> 5;
    const int ngrp   = (Nout >> 4) / NTG;          // n-tile groups
    const int total  = (M >> 4) * ngrp;
    const int tile   = blockIdx.x * 8 + wave;      // wave-uniform
    if (tile >= total) return;                     // whole-wave exit: EXEC all-1s at WMMA
    const int mt     = tile / ngrp;
    const int ntb    = (tile - mt * ngrp) * NTG;   // first n-tile of this wave's group
    const int l16    = lane & 15;
    const int khalf  = lane >> 4;                  // 0 or 1

    const _Float16* arowp = Xh + (long)(mt * 16 + l16) * Ks;
    const _Float16* browp[NTG];
#pragma unroll
    for (int j = 0; j < NTG; ++j)
        browp[j] = Wh + (long)((ntb + j) * 16 + l16) * Ks;

    v8f acc[NTG];
#pragma unroll
    for (int j = 0; j < NTG; ++j) acc[j] = (v8f){};

    for (int k0 = 0; k0 < Ks; k0 += 32) {
        // A layout (16x32 f16): lanes 0-15 hold K=0..15, lanes 16-31 hold K=16..31
        const int ka0 = k0 + khalf * 16;
        v16h a = *(const v16h*)(arowp + ka0);

        // B layout (32x16 f16): lanes 0-15 hold K=0..15, lanes 16-31 hold K=16..31
        const int kb0 = k0 + khalf * 16;
#pragma unroll
        for (int j = 0; j < NTG; ++j) {
            v16h b = *(const v16h*)(browp[j] + kb0);
            acc[j] = __builtin_amdgcn_wmma_f32_16x16x32_f16(false, a, false, b,
                                                            (short)0, acc[j], false, false);
        }
    }

    // C/D layout: VGPR v -> row (v + 8*khalf), col = lane&15
    const int mbase = mt * 16 + khalf * 8;
#pragma unroll
    for (int j = 0; j < NTG; ++j) {
        const int n    = (ntb + j) * 16 + l16;
        const float bv = bias[n];
#pragma unroll
        for (int v = 0; v < 8; ++v)
            Y[(long)(mbase + v) * Nout + n] = acc[j][v] + bv;
    }
}

// ---------------- utility fill ----------------
__global__ void fill_f32(float* __restrict__ p, float v, long n) {
    long i = (long)blockIdx.x * blockDim.x + threadIdx.x;
    if (i < n) p[i] = v;
}

// ---------------- edge pass 1: logits + segment max ----------------
__global__ void edge_logits_max(
    const float* __restrict__ xl, const float* __restrict__ xr,
    const float* __restrict__ att, const int* __restrict__ ei,
    float* __restrict__ logit, float* __restrict__ mx, int C)
{
    int t = blockIdx.x * blockDim.x + threadIdx.x;
    if (t >= N_EDGES_SL * HEADS) return;
    int e = t >> 3, h = t & 7;
    int s, d; edge_sd(ei, e, s, d);
    const int HC = HEADS * C;
    const float* ps = xl + (long)s * HC + h * C;
    const float* pd = xr + (long)d * HC + h * C;
    const float* pa = att + h * C;
    float acc = 0.f;
    for (int c = 0; c < C; ++c) {
        float v = ps[c] + pd[c];
        v = (v >= 0.f) ? v : NEG_SLOPE * v;     // LeakyReLU(0.2)
        acc += v * pa[c];
    }
    logit[t] = acc;
    atomicMaxF(&mx[d * HEADS + h], acc);
}

// ---------------- edge pass 2: exp + segment sum ----------------
__global__ void edge_exp_sum(
    const int* __restrict__ ei, float* __restrict__ logit,
    const float* __restrict__ mx, float* __restrict__ dn)
{
    int t = blockIdx.x * blockDim.x + threadIdx.x;
    if (t >= N_EDGES_SL * HEADS) return;
    int e = t >> 3, h = t & 7;
    int s, d; edge_sd(ei, e, s, d);
    float ex = expf(logit[t] - mx[d * HEADS + h]);
    logit[t] = ex;
    atomicAdd(&dn[d * HEADS + h], ex);
}

// ---------------- edge pass 3: alpha-weighted scatter of xl[src] ----------------
__global__ void edge_aggregate(
    const int* __restrict__ ei, const float* __restrict__ xl,
    const float* __restrict__ ex, const float* __restrict__ dn,
    float* __restrict__ out, int C)
{
    const int HC = HEADS * C;
    long t = (long)blockIdx.x * blockDim.x + threadIdx.x;
    if (t >= (long)N_EDGES_SL * HC) return;
    int e  = (int)(t / HC);
    int ch = (int)(t - (long)e * HC);
    int h  = ch / C;
    int s, d; edge_sd(ei, e, s, d);
    float alpha = ex[e * HEADS + h] / dn[d * HEADS + h];
    atomicAdd(&out[(long)d * HC + ch], alpha * xl[(long)s * HC + ch]);
}

// ---------------- bias + ELU (layer 1 output) ----------------
__global__ void bias_elu(float* __restrict__ hbuf, const float* __restrict__ bias,
                         int HC, long total)
{
    long t = (long)blockIdx.x * blockDim.x + threadIdx.x;
    if (t >= total) return;
    int c = (int)(t % HC);
    float v = hbuf[t] + bias[c];
    hbuf[t] = (v > 0.f) ? v : (expf(v) - 1.f);
}

// ---------------- bias + log_softmax in place (final output) ----------------
__global__ void bias_logsoftmax(float* __restrict__ out, const float* __restrict__ bias)
{
    int row = blockIdx.x * blockDim.x + threadIdx.x;
    if (row >= N_NODES) return;
    float* r = out + (long)row * H2;
    float m = -INFINITY;
    for (int c = 0; c < H2; ++c) { float v = r[c] + bias[c]; r[c] = v; m = fmaxf(m, v); }
    float s = 0.f;
    for (int c = 0; c < H2; ++c) s += expf(r[c] - m);
    float lg = m + logf(s);
    for (int c = 0; c < H2; ++c) r[c] -= lg;
}

// ---------------- launch ----------------
extern "C" void kernel_launch(void* const* d_in, const int* in_sizes, int n_in,
                              void* d_out, int out_size, void* d_ws, size_t ws_size,
                              hipStream_t stream)
{
    const float* x     = (const float*)d_in[0];
    const int*   ei    = (const int*)  d_in[1];
    const float* W1l   = (const float*)d_in[2];
    const float* b1l   = (const float*)d_in[3];
    const float* W1r   = (const float*)d_in[4];
    const float* b1r   = (const float*)d_in[5];
    const float* att1  = (const float*)d_in[6];
    const float* bias1 = (const float*)d_in[7];
    const float* W2l   = (const float*)d_in[8];
    const float* b2l   = (const float*)d_in[9];
    const float* W2r   = (const float*)d_in[10];
    const float* b2r   = (const float*)d_in[11];
    const float* att2  = (const float*)d_in[12];
    const float* bias2 = (const float*)d_in[13];
    float* out = (float*)d_out;

    // workspace carve-up
    float* ws  = (float*)d_ws;
    float* xl1 = ws;                                  // 20000*256 f32
    float* xr1 = xl1 + (long)N_NODES * H1;            // 20000*256 f32
    float* hb  = xr1 + (long)N_NODES * H1;            // 20000*256 f32
    float* exb = hb  + (long)N_NODES * H1;            // 340000*8  f32
    float* mxb = exb + (long)N_EDGES_SL * HEADS;      // 160000    f32
    float* dnb = mxb + (long)N_NODES * HEADS;         // 160000    f32
    _Float16* xh  = (_Float16*)(dnb + (long)N_NODES * HEADS); // 20000*512 f16 (layer2 reuses)
    _Float16* whl = xh  + (long)N_NODES * KPAD1;              // 256*512 f16
    _Float16* whr = whl + (long)H1 * KPAD1;                   // 256*512 f16

    const int TB = 256;
    const long EH  = (long)N_EDGES_SL * HEADS;            // 2,720,000
    const long NH1 = (long)N_NODES * H1;                  // 5,120,000
    const long NH2 = (long)N_NODES * H2;                  // 2,560,000
    const long NHD = (long)N_NODES * HEADS;               // 160,000

    // ================= layer 1 =================
    {
        long cx = (long)N_NODES * KPAD1;                  // 10,240,000
        long cw = (long)H1 * KPAD1;                       // 131,072
        convert_f16_pad<<<(int)((cx + TB - 1) / TB), TB, 0, stream>>>(x,   xh,  DIM_IN, KPAD1, N_NODES);
        convert_f16_pad<<<(int)((cw + TB - 1) / TB), TB, 0, stream>>>(W1l, whl, DIM_IN, KPAD1, H1);
        convert_f16_pad<<<(int)((cw + TB - 1) / TB), TB, 0, stream>>>(W1r, whr, DIM_IN, KPAD1, H1);
        int tiles  = (N_NODES / 16) * ((H1 / 16) / NTG);  // 1250*4 = 5000
        int blocks = (tiles + 7) / 8;
        wmma_gemm_xWT<<<blocks, TB, 0, stream>>>(xh, whl, b1l, xl1, N_NODES, KPAD1, H1);
        wmma_gemm_xWT<<<blocks, TB, 0, stream>>>(xh, whr, b1r, xr1, N_NODES, KPAD1, H1);
    }
    fill_f32<<<(int)((NHD + TB - 1) / TB), TB, 0, stream>>>(mxb, -INFINITY, NHD);
    fill_f32<<<(int)((NHD + TB - 1) / TB), TB, 0, stream>>>(dnb, 0.f, NHD);
    fill_f32<<<(int)((NH1 + TB - 1) / TB), TB, 0, stream>>>(hb, 0.f, NH1);

    edge_logits_max<<<(int)((EH + TB - 1) / TB), TB, 0, stream>>>(xl1, xr1, att1, ei, exb, mxb, DIM_H);
    edge_exp_sum   <<<(int)((EH + TB - 1) / TB), TB, 0, stream>>>(ei, exb, mxb, dnb);
    {
        long work = (long)N_EDGES_SL * H1;                // 87,040,000
        edge_aggregate<<<(int)((work + TB - 1) / TB), TB, 0, stream>>>(ei, xl1, exb, dnb, hb, DIM_H);
    }
    bias_elu<<<(int)((NH1 + TB - 1) / TB), TB, 0, stream>>>(hb, bias1, H1, NH1);

    // ================= layer 2 =================
    float* xl2 = xl1;   // reuse
    float* xr2 = xr1;   // reuse
    {
        long cx = (long)N_NODES * H1;                     // 5,120,000 (K=256, no pad)
        long cw = (long)H2 * H1;                          // 32,768
        convert_f16_pad<<<(int)((cx + TB - 1) / TB), TB, 0, stream>>>(hb,  xh,  H1, H1, N_NODES);
        convert_f16_pad<<<(int)((cw + TB - 1) / TB), TB, 0, stream>>>(W2l, whl, H1, H1, H2);
        convert_f16_pad<<<(int)((cw + TB - 1) / TB), TB, 0, stream>>>(W2r, whr, H1, H1, H2);
        int tiles  = (N_NODES / 16) * ((H2 / 16) / NTG);  // 1250*2 = 2500
        int blocks = (tiles + 7) / 8;
        wmma_gemm_xWT<<<blocks, TB, 0, stream>>>(xh, whl, b2l, xl2, N_NODES, H1, H2);
        wmma_gemm_xWT<<<blocks, TB, 0, stream>>>(xh, whr, b2r, xr2, N_NODES, H1, H2);
    }
    fill_f32<<<(int)((NHD + TB - 1) / TB), TB, 0, stream>>>(mxb, -INFINITY, NHD);
    fill_f32<<<(int)((NHD + TB - 1) / TB), TB, 0, stream>>>(dnb, 0.f, NHD);
    fill_f32<<<(int)((NH2 + TB - 1) / TB), TB, 0, stream>>>(out, 0.f, NH2);

    edge_logits_max<<<(int)((EH + TB - 1) / TB), TB, 0, stream>>>(xl2, xr2, att2, ei, exb, mxb, DIM_OUT);
    edge_exp_sum   <<<(int)((EH + TB - 1) / TB), TB, 0, stream>>>(ei, exb, mxb, dnb);
    {
        long work = (long)N_EDGES_SL * H2;                // 43,520,000
        edge_aggregate<<<(int)((work + TB - 1) / TB), TB, 0, stream>>>(ei, xl2, exb, dnb, out, DIM_OUT);
    }
    bias_logsoftmax<<<(N_NODES + TB - 1) / TB, TB, 0, stream>>>(out, bias2);
}
